// StatefulDurationRegulator_84722524881495
// MI455X (gfx1250) — compile-verified, matrix operand order checked by reading, not run on hardware
//
#include <hip/hip_runtime.h>

typedef __bf16 bf16;
typedef bf16  v16bf __attribute__((ext_vector_type(16)));
typedef bf16  v8bf  __attribute__((ext_vector_type(8)));
typedef bf16  v4bf  __attribute__((ext_vector_type(4)));
typedef float v8f   __attribute__((ext_vector_type(8)));

#define B_    8
#define T_    2048
#define H_    512
#define M_    16384      // B*T
#define MOUT_ 20480      // MAX_TOK*T

// ---------------- workspace layout (float elements) ----------------
static constexpr size_t OFF_FILM = 0;                    // 8*512
static constexpr size_t OFF_DTV  = OFF_FILM + 4096;      // 512
static constexpr size_t OFF_AN   = OFF_DTV  + 512;       // 512*512
static constexpr size_t OFF_YS   = OFF_AN   + 262144;    // 16384
static constexpr size_t OFF_ITOK = OFF_YS   + 16384;     // 16384 (int)
static constexpr size_t OFF_CUM  = OFF_ITOK + 16384;     // 16384 (int)
static constexpr size_t OFF_A0   = OFF_CUM  + 16384;     // 16384*512   x / fusedpre
static constexpr size_t OFF_A1   = OFF_A0   + 8388608;   // 16384*1024  xz / shared
static constexpr size_t OFF_A2   = OFF_A1   + 16777216;  // 16384*512   xc / ctx
static constexpr size_t OFF_A3   = OFF_A2   + 8388608;   // 16384*512   Bseq / g / sf
static constexpr size_t OFF_A4   = OFF_A3   + 8388608;   // 16384*512   Cseq / y / fused
static constexpr size_t OFF_BF   = OFF_A4   + 8388608;   // bf16 weight pool

// bf16-element offsets inside the bf16 pool
static constexpr size_t BWIN  = 0;                       // 512*512
static constexpr size_t BWSSM = BWIN  + 262144;          // 1024*512
static constexpr size_t BWB   = BWSSM + 524288;          // 512*512
static constexpr size_t BWC   = BWB   + 262144;          // 512*512
static constexpr size_t BWOUT = BWC   + 262144;          // 512*512
static constexpr size_t BWFUS = BWOUT + 262144;          // 512*1024
static constexpr size_t BWBK  = BWFUS + 524288;          // 256*512
static constexpr size_t BWP   = BWBK  + 131072;          // 4*3*128*512

// ---------------- helpers ----------------
__device__ __forceinline__ float silu_f(float x)    { return x / (1.0f + __expf(-x)); }
__device__ __forceinline__ float sigmoid_f(float x) { return 1.0f / (1.0f + __expf(-x)); }
__device__ __forceinline__ float softplus_f(float x){ return (x > 20.f) ? x : log1pf(__expf(x)); }

// Per-lane bf16 WMMA fragment from an LDS row: two 16B chunks.
// lanes 0-15: K {0..7,16..23}; lanes 16-31: K {8..15,24..31}; p = &row[(lane>>4)*8]
__device__ __forceinline__ v16bf frag_ld(const bf16* p) {
  v8bf lo = *(const v8bf*)p;
  v8bf hi = *(const v8bf*)(p + 16);
  v16bf f;
#pragma unroll
  for (int i = 0; i < 8; ++i) { f[i] = lo[i]; f[i + 8] = hi[i]; }
  return f;
}

// ---------------- generic WMMA GEMM: C[M,N] = act(A @ W^T) ----------------
// A: [M,lda] fp32 (cols < Ksplit), A2: [M,K-Ksplit] fp32 (cols >= Ksplit)
// W: [N,K] bf16 (pre-converted).  FiLM: scale A by (1+film[row>>11, k]).
template<bool FILM, bool RELU>
__global__ __launch_bounds__(256)
void wmma_gemm_kernel(const float* __restrict__ A, int lda,
                      const float* __restrict__ A2, int Ksplit,
                      const bf16* __restrict__ W,
                      const float* __restrict__ film,
                      float* __restrict__ C, int N, int K)
{
  __shared__ bf16 lA[2][64][40];
  __shared__ bf16 lB[2][64][40];
  const int tid  = threadIdx.x;
  const int lane = tid & 31;
  const int wave = tid >> 5;
  const int wy   = wave >> 1;       // 0..3 : 16-row slice
  const int wx   = wave & 1;        // 0..1 : 32-col slice
  const int m0   = blockIdx.y * 64;
  const int n0   = blockIdx.x * 64;

  const int ra = tid >> 3;          // 0..31  (A rows ra, ra+32)
  const int ca = (tid & 7) * 4;     // 0..28
  const int rb = tid >> 2;          // 0..63  (B row)
  const int cb = (tid & 3) * 8;     // 0,8,16,24

  auto load_tile = [&](int buf, int kk) {
    const float* Asrc; size_t astr; int kb;     // uniform per chunk
    if (kk < Ksplit) { Asrc = A;  astr = (size_t)lda;           kb = kk; }
    else             { Asrc = A2; astr = (size_t)(K - Ksplit);  kb = kk - Ksplit; }
    float4 a0 = *(const float4*)(Asrc + (size_t)(m0 + ra)      * astr + kb + ca);
    float4 a1 = *(const float4*)(Asrc + (size_t)(m0 + ra + 32) * astr + kb + ca);
    v8bf  wv = *(const v8bf*)(W + (size_t)(n0 + rb) * (size_t)K + kk + cb);
    if (FILM) {
      float4 f0 = *(const float4*)(film + (size_t)((m0 + ra)      >> 11) * 512 + kk + ca);
      float4 f1 = *(const float4*)(film + (size_t)((m0 + ra + 32) >> 11) * 512 + kk + ca);
      a0.x *= 1.f + f0.x; a0.y *= 1.f + f0.y; a0.z *= 1.f + f0.z; a0.w *= 1.f + f0.w;
      a1.x *= 1.f + f1.x; a1.y *= 1.f + f1.y; a1.z *= 1.f + f1.z; a1.w *= 1.f + f1.w;
    }
    v4bf q0 = { (bf16)a0.x, (bf16)a0.y, (bf16)a0.z, (bf16)a0.w };
    v4bf q1 = { (bf16)a1.x, (bf16)a1.y, (bf16)a1.z, (bf16)a1.w };
    *(v4bf*)&lA[buf][ra][ca]      = q0;
    *(v4bf*)&lA[buf][ra + 32][ca] = q1;
    *(v8bf*)&lB[buf][rb][cb]      = wv;
  };

  v8f acc0 = {0.f,0.f,0.f,0.f,0.f,0.f,0.f,0.f};
  v8f acc1 = acc0;

  load_tile(0, 0);
  int cur = 0;
  for (int kk = 0; kk < K; kk += 32) {
    __syncthreads();
    if (kk + 32 < K) load_tile(cur ^ 1, kk + 32);
    const bf16* pa  = &lA[cur][wy * 16 + (lane & 15)][(lane >> 4) * 8];
    const bf16* pb0 = &lB[cur][wx * 32 + (lane & 15)][(lane >> 4) * 8];
    v16bf af = frag_ld(pa);
    v16bf b0 = frag_ld(pb0);
    v16bf b1 = frag_ld(pb0 + 16 * 40);
    acc0 = __builtin_amdgcn_wmma_f32_16x16x32_bf16(false, af, false, b0, (short)0, acc0, false, false);
    acc1 = __builtin_amdgcn_wmma_f32_16x16x32_bf16(false, af, false, b1, (short)0, acc1, false, false);
    cur ^= 1;
  }

  const int half = lane >> 4;
  const int nn   = lane & 15;
  const int crow = m0 + wy * 16 + half * 8;
  const int ccol = n0 + wx * 32 + nn;
#pragma unroll
  for (int r = 0; r < 8; ++r) {
    float v0 = acc0[r], v1 = acc1[r];
    if (RELU) { v0 = fmaxf(v0, 0.f); v1 = fmaxf(v1, 0.f); }
    C[(size_t)(crow + r) * (size_t)N + ccol]      = v0;
    C[(size_t)(crow + r) * (size_t)N + ccol + 16] = v1;
  }
}

// ---------------- dilated convs as shifted WMMA GEMMs ----------------
// X: [M,512] fp32; WPB: [4][3][128][512] bf16 packed; CTX: [M,512]
__global__ __launch_bounds__(256)
void wmma_conv_kernel(const float* __restrict__ X,
                      const bf16* __restrict__ WPB,
                      float* __restrict__ CTX)
{
  __shared__ bf16 lA[2][64][40];
  __shared__ bf16 lB[2][64][40];
  const int tid  = threadIdx.x;
  const int lane = tid & 31;
  const int wave = tid >> 5;
  const int wy   = wave >> 1;
  const int wx   = wave & 1;
  const int m0   = blockIdx.y * 64;
  const int nb   = blockIdx.x * 64;      // 0 or 64 within this conv's 128 outputs
  const int ci   = blockIdx.z;           // 0..3
  const int dil  = 1 << ci;              // 1,2,4,8
  const bf16* wp = WPB + (size_t)ci * 3 * 128 * 512;

  const int ra = tid >> 3;
  const int ca = (tid & 7) * 4;
  const int rb = tid >> 2;
  const int cb = (tid & 3) * 8;

  auto load_tile = [&](int buf, int q) {   // q = j*16 + kchunk
    const int j  = q >> 4;
    const int kk = (q & 15) * 32;
    const int sh = (j - 1) * dil;
    const bf16* wj = wp + (size_t)j * 128 * 512;
#pragma unroll
    for (int i = 0; i < 2; ++i) {
      const int r = ra + i * 32;
      const int grow = m0 + r;
      const int t  = grow & (T_ - 1);
      int ts = t + sh;
      const float mk = ((unsigned)ts < (unsigned)T_) ? 1.f : 0.f;
      ts = ts < 0 ? 0 : (ts > T_ - 1 ? T_ - 1 : ts);
      float4 a = *(const float4*)(X + (size_t)(grow - t + ts) * 512 + kk + ca);
      v4bf qv = { (bf16)(a.x * mk), (bf16)(a.y * mk), (bf16)(a.z * mk), (bf16)(a.w * mk) };
      *(v4bf*)&lA[buf][r][ca] = qv;
    }
    *(v8bf*)&lB[buf][rb][cb] = *(const v8bf*)(wj + (size_t)(nb + rb) * 512 + kk + cb);
  };

  v8f acc0 = {0.f,0.f,0.f,0.f,0.f,0.f,0.f,0.f};
  v8f acc1 = acc0;

  load_tile(0, 0);
  int cur = 0;
  for (int q = 0; q < 48; ++q) {
    __syncthreads();
    if (q + 1 < 48) load_tile(cur ^ 1, q + 1);
    const bf16* pa  = &lA[cur][wy * 16 + (lane & 15)][(lane >> 4) * 8];
    const bf16* pb0 = &lB[cur][wx * 32 + (lane & 15)][(lane >> 4) * 8];
    v16bf af = frag_ld(pa);
    v16bf b0 = frag_ld(pb0);
    v16bf b1 = frag_ld(pb0 + 16 * 40);
    acc0 = __builtin_amdgcn_wmma_f32_16x16x32_bf16(false, af, false, b0, (short)0, acc0, false, false);
    acc1 = __builtin_amdgcn_wmma_f32_16x16x32_bf16(false, af, false, b1, (short)0, acc1, false, false);
    cur ^= 1;
  }

  const int half = lane >> 4;
  const int nn   = lane & 15;
  const int crow = m0 + wy * 16 + half * 8;
  const int ccol = ci * 128 + nb + wx * 32 + nn;
#pragma unroll
  for (int r = 0; r < 8; ++r) {
    CTX[(size_t)(crow + r) * 512 + ccol]      = acc0[r];
    CTX[(size_t)(crow + r) * 512 + ccol + 16] = acc1[r];
  }
}

// ---------------- FiLM: film[b,:] = relu(style@W1^T)@W2^T ----------------
__global__ __launch_bounds__(256)
void film_kernel(const float* __restrict__ style, const float* __restrict__ W1,
                 const float* __restrict__ W2, float* __restrict__ film)
{
  __shared__ float hid[512];
  const int b = blockIdx.x, tid = threadIdx.x;
  for (int hh = tid; hh < 512; hh += 256) {
    float s = 0.f;
    for (int k = 0; k < 128; ++k) s = fmaf(style[b * 128 + k], W1[(size_t)hh * 128 + k], s);
    hid[hh] = fmaxf(s, 0.f);
  }
  __syncthreads();
  for (int o = tid; o < 512; o += 256) {
    float s = 0.f;
    for (int k = 0; k < 512; ++k) s = fmaf(hid[k], W2[(size_t)o * 512 + k], s);
    film[b * 512 + o] = s;
  }
}

// ---------------- precompute An = -exp(A_log), dtv = softplus(dt_w) ----------------
__global__ void prep_kernel(const float* __restrict__ Alog, const float* __restrict__ dtw,
                            float* __restrict__ An, float* __restrict__ dtv)
{
  int i = blockIdx.x * blockDim.x + threadIdx.x;
  if (i < 512 * 512) An[i] = -__expf(Alog[i]);
  if (i < 512) dtv[i] = softplus_f(dtw[i]);
}

// ---------------- fp32 -> bf16 weight conversion ----------------
__global__ void cvt_kernel(const float* __restrict__ s, bf16* __restrict__ d, int n)
{
  int i = blockIdx.x * blockDim.x + threadIdx.x;
  if (i < n) d[i] = (bf16)s[i];
}

// ---------------- pack conv weights [O,I,3] -> bf16 [ci][j][O][I] ----------------
__global__ void wpack_kernel(const float* __restrict__ c1, const float* __restrict__ c2,
                             const float* __restrict__ c4, const float* __restrict__ c8,
                             bf16* __restrict__ wp)
{
  int i = blockIdx.x * blockDim.x + threadIdx.x;
  if (i >= 4 * 3 * 128 * 512) return;
  int ci = i / (3 * 128 * 512);
  int r  = i % (3 * 128 * 512);
  int j  = r / (128 * 512);
  int r2 = r % (128 * 512);
  int o  = r2 / 512, k = r2 % 512;
  const float* src = (ci == 0) ? c1 : (ci == 1) ? c2 : (ci == 2) ? c4 : c8;
  wp[i] = (bf16)src[(size_t)o * 1536 + (size_t)k * 3 + j];
}

// ---------------- causal depthwise conv k=4 + SiLU ----------------
__global__ __launch_bounds__(256)
void dwconv_kernel(const float* __restrict__ xz, const float* __restrict__ cw,
                   const float* __restrict__ cb, float* __restrict__ xc)
{
  int i = blockIdx.x * blockDim.x + threadIdx.x;   // M*512
  int r = i >> 9, d = i & 511;
  int t = r & (T_ - 1), base = r - t;
  float s = cb[d];
#pragma unroll
  for (int j = 0; j < 4; ++j) {
    int tj = t - 3 + j;
    if (tj >= 0) s = fmaf(cw[d * 4 + j], xz[(size_t)(base + tj) * 1024 + d], s);
  }
  xc[i] = silu_f(s);
}

// ---------------- sequential SSM scan, one workgroup per batch ----------------
__global__ __launch_bounds__(512)
void scan_kernel(const float* __restrict__ An, const float* __restrict__ dtv,
                 const float* __restrict__ xc, const float* __restrict__ Bs,
                 const float* __restrict__ Cs, const float* __restrict__ h0,
                 float* __restrict__ ys)
{
  __shared__ float h[512];
  __shared__ float redb[16];
  __shared__ float redc[16];
  const int j = threadIdx.x;
  const int lane = j & 31, wv = j >> 5;
  const int b = blockIdx.x;
  h[j] = h0[b * 512 + j];
  const float dtj = dtv[j];
  __syncthreads();
  const size_t rowbase = (size_t)b * T_;
  const float* Acol = An + j;
  for (int t = 0; t < T_; ++t) {
    const size_t off = (rowbase + t) * 512 + j;
    const float u  = xc[off];
    const float Bt = Bs[off];
    const float Ct = Cs[off];
    // sumB across 512 (wave shfl + 16-wave LDS combine)
    float wb = Bt;
#pragma unroll
    for (int o = 16; o; o >>= 1) wb += __shfl_xor(wb, o, 32);
    if (lane == 0) redb[wv] = wb;
    __syncthreads();                                    // (1)
    float sumB = 0.f;
#pragma unroll
    for (int w = 0; w < 16; ++w) sumB += redb[w];
    // matvec: acc = sum_i h[i] * A[i,j]  (coalesced across j)
    float acc = 0.f;
#pragma unroll 8
    for (int i = 0; i < 512; ++i) acc = fmaf(h[i], Acol[(size_t)i * 512], acc);
    const float hn = h[j] + dtj * (acc + u * sumB);
    float wc = Ct * hn;
#pragma unroll
    for (int o = 16; o; o >>= 1) wc += __shfl_xor(wc, o, 32);
    if (lane == 0) redc[wv] = wc;
    __syncthreads();                                    // (2) matvec reads done
    h[j] = hn;
    if (j == 0) {
      float ysc = 0.f;
#pragma unroll
      for (int w = 0; w < 16; ++w) ysc += redc[w];
      ys[rowbase + t] = ysc;
    }
    __syncthreads();                                    // (3) h updated
  }
}

// ---------------- g = (yscal + D_p*u) * silu(x_res) ----------------
__global__ __launch_bounds__(256)
void g_kernel(const float* __restrict__ ys, const float* __restrict__ Dp,
              const float* __restrict__ xc, const float* __restrict__ xz,
              float* __restrict__ g)
{
  int i = blockIdx.x * blockDim.x + threadIdx.x;
  int r = i >> 9, d = i & 511;
  float u  = xc[i];
  float xr = xz[(size_t)r * 1024 + 512 + d];
  g[i] = (ys[r] + Dp[d] * u) * silu_f(xr);
}

// ---------------- row LayerNorm (H=512), wave-per-row ----------------
template<bool RES>
__global__ __launch_bounds__(256)
void rownorm_kernel(const float* __restrict__ X, const float* __restrict__ R,
                    const float* __restrict__ gam, const float* __restrict__ bet,
                    float* __restrict__ O)
{
  const int lane = threadIdx.x & 31;
  const int wave = threadIdx.x >> 5;
  const int row  = blockIdx.x * 8 + wave;
  const float* x = X + (size_t)row * 512;
  float v[16];
  float s = 0.f, sq = 0.f;
#pragma unroll
  for (int i = 0; i < 16; ++i) {
    int c = lane + 32 * i;
    float t = x[c];
    if (RES) t += R[(size_t)row * 512 + c];
    v[i] = t; s += t; sq += t * t;
  }
#pragma unroll
  for (int off = 16; off; off >>= 1) { s += __shfl_xor(s, off, 32); sq += __shfl_xor(sq, off, 32); }
  const float mean = s * (1.f / 512.f);
  const float var  = sq * (1.f / 512.f) - mean * mean;
  const float inv  = rsqrtf(var + 1e-5f);
  float* o = O + (size_t)row * 512;
#pragma unroll
  for (int i = 0; i < 16; ++i) {
    int c = lane + 32 * i;
    o[c] = (v[i] - mean) * inv * gam[c] + bet[c];
  }
}

// ---------------- heads: dur/conf/tokens, wave-per-row ----------------
__global__ __launch_bounds__(256)
void heads_kernel(const float* __restrict__ SH, const float* __restrict__ Wd,
                  const float* __restrict__ Wc, float* __restrict__ dur,
                  float* __restrict__ tokf, float* __restrict__ conf,
                  int* __restrict__ itok)
{
  const int lane = threadIdx.x & 31;
  const int wave = threadIdx.x >> 5;
  const int row  = blockIdx.x * 8 + wave;
  const float* s = SH + (size_t)row * 256;
  float sd = 0.f, sc = 0.f;
#pragma unroll
  for (int i = 0; i < 8; ++i) {
    int c = lane + 32 * i;
    float t = s[c];
    sd = fmaf(t, Wd[c], sd);
    sc = fmaf(t, Wc[c], sc);
  }
#pragma unroll
  for (int off = 16; off; off >>= 1) { sd += __shfl_xor(sd, off, 32); sc += __shfl_xor(sc, off, 32); }
  if (lane == 0) {
    float d  = softplus_f(sd);
    float cf = sigmoid_f(sc);
    int tk = (int)rintf(d * 75.0f);
    tk = tk < 1 ? 1 : (tk > 10 ? 10 : tk);
    dur[row] = d; conf[row] = cf; tokf[row] = (float)tk; itok[row] = tk;
  }
}

// ---------------- per-batch cumulative sum of tokens ----------------
__global__ void cum_kernel(const int* __restrict__ itok, int* __restrict__ cum)
{
  int b = threadIdx.x;
  if (b >= B_) return;
  int run = 0;
  for (int t = 0; t < T_; ++t) { run += itok[b * T_ + t]; cum[b * T_ + t] = run; }
}

// ---------------- length regulation: binary search + float4 row stream ----------------
__global__ __launch_bounds__(256)
void regulate_kernel(const float* __restrict__ text, const int* __restrict__ cum,
                     float* __restrict__ out)
{
  const int sub = threadIdx.x >> 7;         // row within block
  const int lr  = threadIdx.x & 127;        // float4 index within row
  const int rid = blockIdx.x * 2 + sub;     // 0 .. B*MOUT-1
  const int b = rid / MOUT_;
  const int m = rid % MOUT_;
  const int* c = cum + b * T_;
  const int total = c[T_ - 1];
  int lo = 0, hi = T_;
  while (lo < hi) { int mid = (lo + hi) >> 1; if (c[mid] <= m) lo = mid + 1; else hi = mid; }
  const int idx = lo > (T_ - 1) ? (T_ - 1) : lo;
  const float4* src = (const float4*)(text + ((size_t)b * T_ + idx) * 512);
  float4* dst = (float4*)(out + ((size_t)b * MOUT_ + m) * 512);
  float4 z = {0.f, 0.f, 0.f, 0.f};
  dst[lr] = (m < total) ? src[lr] : z;
}

// ---------------- launch ----------------
extern "C" void kernel_launch(void* const* d_in, const int* in_sizes, int n_in,
                              void* d_out, int out_size, void* d_ws, size_t ws_size,
                              hipStream_t stream)
{
  const float* text  = (const float*)d_in[0];
  const float* style = (const float*)d_in[1];
  const float* h0    = (const float*)d_in[2];
  const float* Wst1  = (const float*)d_in[3];
  const float* Wst2  = (const float*)d_in[4];
  const float* Win   = (const float*)d_in[5];
  const float* Wssm  = (const float*)d_in[6];
  const float* convw = (const float*)d_in[7];
  const float* convb = (const float*)d_in[8];
  const float* dtw   = (const float*)d_in[9];
  const float* Alog  = (const float*)d_in[10];
  const float* Dp    = (const float*)d_in[11];
  const float* WB    = (const float*)d_in[12];
  const float* WC    = (const float*)d_in[13];
  const float* Wout  = (const float*)d_in[14];
  const float* ln1g  = (const float*)d_in[15];
  const float* ln1b  = (const float*)d_in[16];
  const float* cw1   = (const float*)d_in[17];
  const float* cw2   = (const float*)d_in[18];
  const float* cw4   = (const float*)d_in[19];
  const float* cw8   = (const float*)d_in[20];
  const float* Wfuse = (const float*)d_in[21];
  const float* ln2g  = (const float*)d_in[22];
  const float* ln2b  = (const float*)d_in[23];
  const float* Wback = (const float*)d_in[24];
  const float* Wdur  = (const float*)d_in[25];
  const float* Wconf = (const float*)d_in[26];
  (void)in_sizes; (void)n_in; (void)out_size; (void)ws_size;

  float* ws   = (float*)d_ws;
  float* film = ws + OFF_FILM;
  float* dtv  = ws + OFF_DTV;
  float* An   = ws + OFF_AN;
  float* ys   = ws + OFF_YS;
  int*   itok = (int*)(ws + OFF_ITOK);
  int*   cum  = (int*)(ws + OFF_CUM);
  float* x    = ws + OFF_A0;   // x, later fusedpre
  float* xz   = ws + OFF_A1;   // xz, later shared
  float* xc   = ws + OFF_A2;   // xc, later ctx
  float* bs   = ws + OFF_A3;   // Bseq, later g, later sf
  float* cs   = ws + OFF_A4;   // Cseq, later y, later fused
  bf16*  bfp  = (bf16*)(ws + OFF_BF);

  float* out  = (float*)d_out;
  float* dur  = out + (size_t)B_ * MOUT_ * 512;
  float* tokf = dur + M_;
  float* conf = tokf + M_;

  // one-shot precompute: FiLM, -exp(A), softplus(dt), bf16 weight pool
  film_kernel<<<B_, 256, 0, stream>>>(style, Wst1, Wst2, film);
  prep_kernel<<<1024, 256, 0, stream>>>(Alog, dtw, An, dtv);
  cvt_kernel<<<1024, 256, 0, stream>>>(Win,   bfp + BWIN,  262144);
  cvt_kernel<<<2048, 256, 0, stream>>>(Wssm,  bfp + BWSSM, 524288);
  cvt_kernel<<<1024, 256, 0, stream>>>(WB,    bfp + BWB,   262144);
  cvt_kernel<<<1024, 256, 0, stream>>>(WC,    bfp + BWC,   262144);
  cvt_kernel<<<1024, 256, 0, stream>>>(Wout,  bfp + BWOUT, 262144);
  cvt_kernel<<<2048, 256, 0, stream>>>(Wfuse, bfp + BWFUS, 524288);
  cvt_kernel<<<512,  256, 0, stream>>>(Wback, bfp + BWBK,  131072);
  wpack_kernel<<<(4 * 3 * 128 * 512 + 255) / 256, 256, 0, stream>>>(cw1, cw2, cw4, cw8, bfp + BWP);

  // x = (text * (1+film)) @ W_in^T
  wmma_gemm_kernel<true, false><<<dim3(8, M_ / 64), 256, 0, stream>>>(
      text, 512, text, 512, bfp + BWIN, film, x, 512, 512);
  // xz = x @ W_ssm_in^T   [M,1024]
  wmma_gemm_kernel<false, false><<<dim3(16, M_ / 64), 256, 0, stream>>>(
      x, 512, x, 512, bfp + BWSSM, nullptr, xz, 1024, 512);
  // xc = silu(causal dwconv(x_in))
  dwconv_kernel<<<(M_ * 512) / 256, 256, 0, stream>>>(xz, convw, convb, xc);
  // Bseq, Cseq
  wmma_gemm_kernel<false, false><<<dim3(8, M_ / 64), 256, 0, stream>>>(
      xc, 512, xc, 512, bfp + BWB, nullptr, bs, 512, 512);
  wmma_gemm_kernel<false, false><<<dim3(8, M_ / 64), 256, 0, stream>>>(
      xc, 512, xc, 512, bfp + BWC, nullptr, cs, 512, 512);
  // sequential scan -> ys[b,t]
  scan_kernel<<<B_, 512, 0, stream>>>(An, dtv, xc, bs, cs, h0, ys);
  // g -> bs (Bseq dead)
  g_kernel<<<(M_ * 512) / 256, 256, 0, stream>>>(ys, Dp, xc, xz, bs);
  // y = g @ W_out^T -> cs (Cseq dead)
  wmma_gemm_kernel<false, false><<<dim3(8, M_ / 64), 256, 0, stream>>>(
      bs, 512, bs, 512, bfp + BWOUT, nullptr, cs, 512, 512);
  // ctx -> xc (xc dead after g)
  wmma_conv_kernel<<<dim3(2, M_ / 64, 4), 256, 0, stream>>>(x, bfp + BWP, xc);
  // sf = ln1(y + x) -> bs (g dead)
  rownorm_kernel<true><<<M_ / 8, 256, 0, stream>>>(cs, x, ln1g, ln1b, bs);
  // fusedpre = relu(concat(sf, ctx) @ W_fuse^T) -> x (x dead)
  wmma_gemm_kernel<false, true><<<dim3(8, M_ / 64), 256, 0, stream>>>(
      bs, 512, xc, 512, bfp + BWFUS, nullptr, x, 512, 1024);
  // fused = ln2(fusedpre) -> cs
  rownorm_kernel<false><<<M_ / 8, 256, 0, stream>>>(x, nullptr, ln2g, ln2b, cs);
  // shared = relu(fused @ W_back^T) [M,256] -> xz (dead)
  wmma_gemm_kernel<false, true><<<dim3(4, M_ / 64), 256, 0, stream>>>(
      cs, 512, cs, 512, bfp + BWBK, nullptr, xz, 256, 512);
  // heads
  heads_kernel<<<M_ / 8, 256, 0, stream>>>(xz, Wdur, Wconf, dur, tokf, conf, itok);
  // cumsum per batch
  cum_kernel<<<1, 8, 0, stream>>>(itok, cum);
  // length regulation
  regulate_kernel<<<(B_ * MOUT_) / 2, 256, 0, stream>>>(text, cum, out);
}
